// GCNRegressor_83305185673705
// MI455X (gfx1250) — compile-verified, matrix operand order checked by reading, not run on hardware
//
#include <hip/hip_runtime.h>
#include <hip/hip_bf16.h>
#include <stdint.h>

#define N_NODES 100000
#define N_EDGES 1600000
#define IN_CH   128
#define HID     256
#define NUM_GRAPHS 128

typedef __attribute__((ext_vector_type(8)))  __bf16 v8bf;
typedef __attribute__((ext_vector_type(16))) __bf16 v16bf;
typedef __attribute__((ext_vector_type(4)))  __bf16 v4bf;
typedef __attribute__((ext_vector_type(8)))  float  v8f;

// ---------------- utility kernels ----------------

__global__ void k_fill_f32(float* __restrict__ p, float v, int n) {
    int i = blockIdx.x * blockDim.x + threadIdx.x;
    if (i < n) p[i] = v;
}

// degree accumulate: isd pre-filled with 1.0 (self loop)
__global__ void k_deg(const int* __restrict__ dst, float* __restrict__ isd, int e) {
    int i = blockIdx.x * blockDim.x + threadIdx.x;
    if (i < e) atomicAdd(&isd[dst[i]], 1.0f);
}

__global__ void k_rsqrt(float* __restrict__ p, int n) {
    int i = blockIdx.x * blockDim.x + threadIdx.x;
    if (i < n) p[i] = __frsqrt_rn(p[i]);
}

__global__ void k_cvt_bf16(const float* __restrict__ src, __bf16* __restrict__ dst, int n) {
    int i = blockIdx.x * blockDim.x + threadIdx.x;
    if (i < n) dst[i] = (__bf16)src[i];
}

// WT[n*K + k] = bf16(W[k*N + n]) : pack weight transposed for contiguous B-fragment loads
__global__ void k_packWT(const float* __restrict__ W, __bf16* __restrict__ WT,
                         int K, int Ncols) {
    int i = blockIdx.x * blockDim.x + threadIdx.x;
    if (i < K * Ncols) {
        int k = i / Ncols, n = i % Ncols;
        WT[n * K + k] = (__bf16)W[i];
    }
}

// ---------------- WMMA GEMM: C[M,Nc] = A[M,K] (bf16) @ WT[Nc,K]^T (bf16), f32 out ----------
// grid: (ceil(M/128), Nc/64), block: 256 threads = 8 waves.
// Each block: 8 M-tiles x one 64-column slab. B slab async-DMA'd into LDS (shared by waves).
__global__ void __launch_bounds__(256)
k_gemm_bf16(const __bf16* __restrict__ A, const __bf16* __restrict__ BT,
            float* __restrict__ C, int M, int K, int Ncols) {
    __shared__ __bf16 sB[64 * 256];   // 32 KB max (K<=256)

    const int tid  = threadIdx.x;
    const int lane = tid & 31;
    const int wv   = tid >> 5;               // 0..7
    const int g    = lane >> 4;              // K-half group
    const int mn   = lane & 15;              // row (A) / col (B,C) within tile

    // async DMA of 64xK bf16 weight slab (contiguous in BT) straight into LDS,
    // bypassing VGPRs; tracked by ASYNCcnt.
    {
        const __bf16* gbase = BT + (size_t)blockIdx.y * 64 * K;
        const unsigned ldsBase = (unsigned)(uintptr_t)(&sB[0]);  // low 32 bits = LDS offset
        int nchunks = (64 * K) / 8;          // 8 bf16 (16B) per async op
        for (int c = tid; c < nchunks; c += 256) {
            unsigned ldsOff = ldsBase + (unsigned)c * 16u;
            unsigned long long ga = (unsigned long long)(uintptr_t)(gbase + (size_t)c * 8);
            asm volatile("global_load_async_to_lds_b128 %0, %1, off"
                         :: "v"(ldsOff), "v"(ga) : "memory");
        }
        asm volatile("s_wait_asynccnt 0x0" ::: "memory");
    }
    __syncthreads();

    const int mtile = blockIdx.x * 8 + wv;
    if (mtile * 16 >= M) return;             // wave-uniform: EXEC stays all-1s

    const __bf16* arow = A + (size_t)(mtile * 16 + mn) * K + g * 8;
    const __bf16* brow = sB + (size_t)mn * K + g * 8;

    v8f acc0 = {}, acc1 = {}, acc2 = {}, acc3 = {};

    for (int k = 0; k < K; k += 32) {
        // A fragment (two b128 global loads)
        v8bf alo = *(const v8bf*)(arow + k);
        v8bf ahi = *(const v8bf*)(arow + k + 16);
        // all four B fragments up front (eight clause-able ds_load_b128)
        const __bf16* b0 = brow + k;
        v8bf b0lo = *(const v8bf*)(b0 + 0 * 16 * K);
        v8bf b0hi = *(const v8bf*)(b0 + 0 * 16 * K + 16);
        v8bf b1lo = *(const v8bf*)(b0 + 1 * 16 * K);
        v8bf b1hi = *(const v8bf*)(b0 + 1 * 16 * K + 16);
        v8bf b2lo = *(const v8bf*)(b0 + 2 * 16 * K);
        v8bf b2hi = *(const v8bf*)(b0 + 2 * 16 * K + 16);
        v8bf b3lo = *(const v8bf*)(b0 + 3 * 16 * K);
        v8bf b3hi = *(const v8bf*)(b0 + 3 * 16 * K + 16);

        v16bf af  = __builtin_shufflevector(alo,  ahi,  0,1,2,3,4,5,6,7,8,9,10,11,12,13,14,15);
        v16bf bf0 = __builtin_shufflevector(b0lo, b0hi, 0,1,2,3,4,5,6,7,8,9,10,11,12,13,14,15);
        v16bf bf1 = __builtin_shufflevector(b1lo, b1hi, 0,1,2,3,4,5,6,7,8,9,10,11,12,13,14,15);
        v16bf bf2 = __builtin_shufflevector(b2lo, b2hi, 0,1,2,3,4,5,6,7,8,9,10,11,12,13,14,15);
        v16bf bf3 = __builtin_shufflevector(b3lo, b3hi, 0,1,2,3,4,5,6,7,8,9,10,11,12,13,14,15);

        // back-to-back matrix ops (independent accumulators)
        acc0 = __builtin_amdgcn_wmma_f32_16x16x32_bf16(false, af, false, bf0,
                                                       (short)0, acc0, false, false);
        acc1 = __builtin_amdgcn_wmma_f32_16x16x32_bf16(false, af, false, bf1,
                                                       (short)0, acc1, false, false);
        acc2 = __builtin_amdgcn_wmma_f32_16x16x32_bf16(false, af, false, bf2,
                                                       (short)0, acc2, false, false);
        acc3 = __builtin_amdgcn_wmma_f32_16x16x32_bf16(false, af, false, bf3,
                                                       (short)0, acc3, false, false);
    }

    const int ncolbase = blockIdx.y * 64 + mn;
    float* crow = C + (size_t)(mtile * 16 + 8 * g) * Ncols + ncolbase;
    #pragma unroll
    for (int r = 0; r < 8; r++) {
        crow[(size_t)r * Ncols + 0 * 16] = acc0[r];
        crow[(size_t)r * Ncols + 1 * 16] = acc1[r];
        crow[(size_t)r * Ncols + 2 * 16] = acc2[r];
        crow[(size_t)r * Ncols + 3 * 16] = acc3[r];
    }
}

// ---------------- edge scatter: acc[dst] += hw[src] * isd[src]*isd[dst] -----------
// 64 lanes per edge, float4 per lane (256 channels)
__global__ void k_scatter(const float* __restrict__ hw,
                          const int* __restrict__ src, const int* __restrict__ dst,
                          const float* __restrict__ isd, float* __restrict__ acc,
                          int e) {
    long long gid = (long long)blockIdx.x * blockDim.x + threadIdx.x;
    int ed = (int)(gid >> 6);
    if (ed >= e) return;
    int c0 = ((int)gid & 63) * 4;
    int s = src[ed], d = dst[ed];
    float norm = isd[s] * isd[d];
    const float4 v = *(const float4*)(hw + (size_t)s * HID + c0);
    float* a = acc + (size_t)d * HID + c0;
    atomicAdd(a + 0, v.x * norm);
    atomicAdd(a + 1, v.y * norm);
    atomicAdd(a + 2, v.z * norm);
    atomicAdd(a + 3, v.w * norm);
}

// ---------------- epilogue: h = relu(acc + hw*isd^2 + b); write f32 (in hw) + bf16 ----
__global__ void k_epilogue(const float* __restrict__ acc, float* __restrict__ hw,
                           const float* __restrict__ isd, const float* __restrict__ b,
                           __bf16* __restrict__ hbf, int n) {
    int gid = blockIdx.x * blockDim.x + threadIdx.x;   // n = N*HID/4
    if (gid >= n) return;
    int i  = gid >> 6;               // node
    int c0 = (gid & 63) * 4;
    float self = isd[i] * isd[i];
    const float4 a = *(const float4*)(acc + (size_t)i * HID + c0);
    float4 h = *(const float4*)(hw + (size_t)i * HID + c0);
    const float4 bb = *(const float4*)(b + c0);
    h.x = fmaxf(a.x + h.x * self + bb.x, 0.0f);
    h.y = fmaxf(a.y + h.y * self + bb.y, 0.0f);
    h.z = fmaxf(a.z + h.z * self + bb.z, 0.0f);
    h.w = fmaxf(a.w + h.w * self + bb.w, 0.0f);
    *(float4*)(hw + (size_t)i * HID + c0) = h;
    v4bf hb; hb.x = (__bf16)h.x; hb.y = (__bf16)h.y; hb.z = (__bf16)h.z; hb.w = (__bf16)h.w;
    *(v4bf*)(hbf + (size_t)i * HID + c0) = hb;
}

// ---------------- pooling ----------------
__global__ void k_counts(const int* __restrict__ batch, float* __restrict__ cnt, int n) {
    int i = blockIdx.x * blockDim.x + threadIdx.x;
    if (i < n) atomicAdd(&cnt[batch[i]], 1.0f);
}

__global__ void k_pool(const float* __restrict__ h, const int* __restrict__ batch,
                       float* __restrict__ sums, int n) {
    int gid = blockIdx.x * blockDim.x + threadIdx.x;   // n = N*HID/4
    if (gid >= n) return;
    int i  = gid >> 6;
    int c0 = (gid & 63) * 4;
    int gph = batch[i];
    const float4 v = *(const float4*)(h + (size_t)i * HID + c0);
    float* s = sums + (size_t)gph * HID + c0;
    atomicAdd(s + 0, v.x);
    atomicAdd(s + 1, v.y);
    atomicAdd(s + 2, v.z);
    atomicAdd(s + 3, v.w);
}

// ---------------- MLP head: out[g] = relu(hg@Wh1+bh1)@Wh2 + bh2 ----------------
// one block per graph, 128 threads (one per hidden unit)
__global__ void __launch_bounds__(128)
k_head(const float* __restrict__ sums, const float* __restrict__ cnt,
       const float* __restrict__ Wh1, const float* __restrict__ bh1,
       const float* __restrict__ Wh2, const float* __restrict__ bh2,
       float* __restrict__ out) {
    __shared__ float red[128];
    int g = blockIdx.x, j = threadIdx.x;
    float inv = 1.0f / fmaxf(cnt[g], 1.0f);
    float a = bh1[j];
    const float* hg = sums + (size_t)g * HID;
    #pragma unroll 4
    for (int c = 0; c < HID; c++) a += (hg[c] * inv) * Wh1[c * 128 + j];
    red[j] = fmaxf(a, 0.0f) * Wh2[j];
    __syncthreads();
    for (int st = 64; st > 0; st >>= 1) {
        if (j < st) red[j] += red[j + st];
        __syncthreads();
    }
    if (j == 0) out[g] = red[0] + bh2[0];
}

// ---------------- driver ----------------
extern "C" void kernel_launch(void* const* d_in, const int* in_sizes, int n_in,
                              void* d_out, int out_size, void* d_ws, size_t ws_size,
                              hipStream_t stream) {
    const float* x     = (const float*)d_in[0];
    const int*   ei    = (const int*)d_in[1];
    const int*   srcI  = ei;
    const int*   dstI  = ei + N_EDGES;
    const int*   batch = (const int*)d_in[2];
    const float* W1 = (const float*)d_in[3];  const float* b1 = (const float*)d_in[4];
    const float* W2 = (const float*)d_in[5];  const float* b2 = (const float*)d_in[6];
    const float* W3 = (const float*)d_in[7];  const float* b3 = (const float*)d_in[8];
    const float* Wh1 = (const float*)d_in[9]; const float* bh1 = (const float*)d_in[10];
    const float* Wh2 = (const float*)d_in[11];const float* bh2 = (const float*)d_in[12];
    float* out = (float*)d_out;

    // workspace carve-up (aligned to 256B)
    char* ws = (char*)d_ws;
    size_t off = 0;
    auto carve = [&](size_t bytes) { char* p = ws + off; off = (off + bytes + 255) & ~(size_t)255; return p; };
    float*  isd  = (float*) carve((size_t)N_NODES * sizeof(float));
    __bf16* hbf  = (__bf16*)carve((size_t)N_NODES * HID * sizeof(__bf16));
    float*  hw   = (float*) carve((size_t)N_NODES * HID * sizeof(float));
    float*  acc  = (float*) carve((size_t)N_NODES * HID * sizeof(float));
    __bf16* WT   = (__bf16*)carve((size_t)HID * HID * sizeof(__bf16));
    float*  sums = (float*) carve((size_t)NUM_GRAPHS * HID * sizeof(float));
    float*  cnts = (float*) carve((size_t)NUM_GRAPHS * sizeof(float));

    const int T = 256;
    auto blocks = [](long long n, int t) { return (int)((n + t - 1) / t); };

    // normalization: isd = rsqrt(1 + in-degree)
    k_fill_f32<<<blocks(N_NODES, T), T, 0, stream>>>(isd, 1.0f, N_NODES);
    k_deg<<<blocks(N_EDGES, T), T, 0, stream>>>(dstI, isd, N_EDGES);
    k_rsqrt<<<blocks(N_NODES, T), T, 0, stream>>>(isd, N_NODES);

    // input features -> bf16
    k_cvt_bf16<<<blocks((long long)N_NODES * IN_CH, T), T, 0, stream>>>(x, hbf, N_NODES * IN_CH);

    const float* Ws[3] = {W1, W2, W3};
    const float* bs[3] = {b1, b2, b3};
    const int    Ks[3] = {IN_CH, HID, HID};

    const long long nhid4 = (long long)N_NODES * HID / 4;
    for (int L = 0; L < 3; L++) {
        int K = Ks[L];
        k_packWT<<<blocks((long long)K * HID, T), T, 0, stream>>>(Ws[L], WT, K, HID);
        k_fill_f32<<<blocks((long long)N_NODES * HID, T), T, 0, stream>>>(acc, 0.0f, N_NODES * HID);
        dim3 grid((N_NODES / 16 + 7) / 8, HID / 64);
        k_gemm_bf16<<<grid, 256, 0, stream>>>(hbf, WT, hw, N_NODES, K, HID);
        k_scatter<<<blocks((long long)N_EDGES * 64, T), T, 0, stream>>>(hw, srcI, dstI, isd, acc, N_EDGES);
        k_epilogue<<<blocks(nhid4, T), T, 0, stream>>>(acc, hw, isd, bs[L], hbf, (int)nhid4);
    }

    // global mean pool + head
    k_fill_f32<<<blocks(NUM_GRAPHS * HID, T), T, 0, stream>>>(sums, 0.0f, NUM_GRAPHS * HID);
    k_fill_f32<<<1, T, 0, stream>>>(cnts, 0.0f, NUM_GRAPHS);
    k_counts<<<blocks(N_NODES, T), T, 0, stream>>>(batch, cnts, N_NODES);
    k_pool<<<blocks(nhid4, T), T, 0, stream>>>(hw, batch, sums, (int)nhid4);
    k_head<<<NUM_GRAPHS, 128, 0, stream>>>(sums, cnts, Wh1, bh1, Wh2, bh2, out);
}
